// MLGCN_SE_3083786519089
// MI455X (gfx1250) — matrix operand order repaired and checked by reading, not compiled, not run on hardware
//
#include <hip/hip_runtime.h>
#include <hip/hip_bf16.h>
#include <math.h>

#define B_ 128
#define T_ 128
#define DE_ 300
#define DP_ 30
#define DIN_ 330
#define DINP_ 352
#define H_ 300
#define HP_ 320
#define G4_ 1200
#define G4P_ 1216
#define D2_ 600
#define D2P_ 640
#define HEADS_ 6
#define DK_ 100
#define DKP_ 128
#define LA_ 5
#define LL_ 24
#define POL_ 3
#define NEGV (-1000000000.0f)

typedef __bf16 bf16_t;
typedef __bf16 v16bf __attribute__((ext_vector_type(16)));
typedef float  v8f   __attribute__((ext_vector_type(8)));

// ====================== CDNA5 WMMA tile loaders (wave32) ======================
// Contiguous k order (k0 + 16*(lane/16) + e), the same in the A and B loaders.

// A operand from (M,K) row-major. 16x32 bf16 tile.
// lanes 0-15 : M=lane,    K=0..15
// lanes16-31 : M=lane-16, K=16..31
__device__ __forceinline__ v16bf load_a16(const bf16_t* __restrict__ p, int ld,
                                          int row0, int k0, int lane) {
  int r = row0 + (lane & 15);
  int g = (lane >> 4) & 1;
  const bf16_t* q = p + (size_t)r * ld + k0 + g * 16;
  v16bf v;
#pragma unroll
  for (int e = 0; e < 16; ++e) v[e] = q[e];
  return v;
}

// B operand from (N,K) row-major storage (weight [out,in] or keys [T,dk]).
// 32x16 tile: lane n holds column n; lanes 0-15 K=0..15, lanes 16-31 K=16..31.
__device__ __forceinline__ v16bf load_b_nk(const bf16_t* __restrict__ p, int ld,
                                           int k0, int col0, int lane) {
  int n = col0 + (lane & 15);
  int g = (lane >> 4) & 1;
  const bf16_t* q = p + (size_t)n * ld + k0 + g * 16;
  v16bf v;
#pragma unroll
  for (int e = 0; e < 16; ++e) v[e] = q[e];
  return v;
}

// B operand from (K,N) row-major storage (activation used as RHS).
__device__ __forceinline__ v16bf load_b_kn(const bf16_t* __restrict__ p, int ld,
                                           int k0, int col0, int lane) {
  int c = col0 + (lane & 15);
  int g = (lane >> 4) & 1;
  v16bf v;
#pragma unroll
  for (int e = 0; e < 16; ++e) v[e] = p[(size_t)(k0 + g*16 + e) * ld + c];
  return v;
}

// ====================== generic WMMA GEMMs (16x64 macro-tile per wave) =======

// C(M,Nreal) = act( A(M,Kpad)bf16 * W(Npad,Kpad)bf16^T + bias + add )
// One wave computes 1 M-tile x 4 N-tiles: the A fragment is loaded once and
// feeds 4 v_wmma issues (4x less A traffic). Npad must be a multiple of 64.
__global__ void gemm_nk(const bf16_t* __restrict__ A, int lda,
                        const bf16_t* __restrict__ W, int ldb,
                        float* __restrict__ C, int ldc,
                        const float* __restrict__ bias,
                        const float* __restrict__ add, int addStride,
                        int M, int Npad, int Nreal, int Kpad, int relu) {
  int lane  = threadIdx.x & 31;
  int wv    = threadIdx.x >> 5;
  int group = blockIdx.x * 4 + wv;
  int ng = Npad >> 6;          // groups of 4 N-tiles
  int mt = M >> 4;
  if (group >= mt * ng) return;
  int gm = group / ng, gn = group % ng;
  int n0 = gn * 64;
  v8f acc[4];
#pragma unroll
  for (int j = 0; j < 4; ++j) acc[j] = (v8f){0.f,0.f,0.f,0.f,0.f,0.f,0.f,0.f};
  for (int k0 = 0; k0 < Kpad; k0 += 32) {
    if (k0 + 32 < Kpad) {  // global_prefetch_b8 of next K slice
      __builtin_prefetch((const void*)(A + (size_t)(gm*16 + (lane & 15)) * lda + k0 + 32), 0, 1);
      __builtin_prefetch((const void*)(W + (size_t)(n0 + (lane & 15)) * ldb + k0 + 32), 0, 1);
    }
    v16bf a = load_a16(A, lda, gm*16, k0, lane);
#pragma unroll
    for (int j = 0; j < 4; ++j) {
      v16bf b = load_b_nk(W, ldb, k0, n0 + j*16, lane);
      acc[j] = __builtin_amdgcn_wmma_f32_16x16x32_bf16(false, a, false, b,
                                                       (short)0, acc[j], false, false);
    }
  }
  int g = (lane >> 4) & 1;
#pragma unroll
  for (int j = 0; j < 4; ++j) {
    int col = n0 + j*16 + (lane & 15);
    if (col < Nreal) {
      float bv = bias ? bias[col] : 0.f;
#pragma unroll
      for (int r = 0; r < 8; ++r) {
        int row = gm*16 + r + 8*g;
        float v = acc[j][r] + bv;
        if (add)  v += add[(size_t)row * addStride + col];
        if (relu) v = fmaxf(v, 0.f);
        C[(size_t)row * ldc + col] = v;
      }
    }
  }
}

// Per-batch: C[b](T,Nreal) = A[b](T,Kpad)bf16 * X[b](Kpad,Npad)bf16
__global__ void gemm_kn_batched(const bf16_t* __restrict__ A, int lda, long long strideA,
                                const bf16_t* __restrict__ X, int ldx, long long strideX,
                                float* __restrict__ C, int ldc, long long strideC,
                                int Npad, int Nreal, int Kpad) {
  int lane  = threadIdx.x & 31;
  int wv    = threadIdx.x >> 5;
  int group = blockIdx.x * 4 + wv;
  int ng  = Npad >> 6;
  int per = (T_ >> 4) * ng;
  int b = group / per;
  if (b >= B_) return;
  int rest = group % per;
  int gm = rest / ng, gn = rest % ng;
  int n0 = gn * 64;
  const bf16_t* Ab = A + (size_t)b * strideA;
  const bf16_t* Xb = X + (size_t)b * strideX;
  v8f acc[4];
#pragma unroll
  for (int j = 0; j < 4; ++j) acc[j] = (v8f){0.f,0.f,0.f,0.f,0.f,0.f,0.f,0.f};
  for (int k0 = 0; k0 < Kpad; k0 += 32) {
    v16bf a = load_a16(Ab, lda, gm*16, k0, lane);
#pragma unroll
    for (int j = 0; j < 4; ++j) {
      v16bf bb = load_b_kn(Xb, ldx, k0, n0 + j*16, lane);
      acc[j] = __builtin_amdgcn_wmma_f32_16x16x32_bf16(false, a, false, bb,
                                                       (short)0, acc[j], false, false);
    }
  }
  int g = (lane >> 4) & 1;
  float* Cb = C + (size_t)b * strideC;
#pragma unroll
  for (int j = 0; j < 4; ++j) {
    int col = n0 + j*16 + (lane & 15);
    if (col < Nreal) {
#pragma unroll
      for (int r = 0; r < 8; ++r) {
        int row = gm*16 + r + 8*g;
        Cb[(size_t)row * ldc + col] = acc[j][r];
      }
    }
  }
}

// scores[b,h] = (q[b,h](T,128) * k[b,h]^T) * 0.1, key-mask -> NEG.
// 8 M-tiles x 2 groups of 4 N-tiles per (b,h); q fragment reused 4x.
__global__ void scores_gemm(const bf16_t* __restrict__ qh,
                            const bf16_t* __restrict__ kh,
                            float* __restrict__ scores,
                            const int* __restrict__ text_len) {
  int lane  = threadIdx.x & 31;
  int wv    = threadIdx.x >> 5;
  int group = blockIdx.x * 4 + wv;
  if (group >= B_ * HEADS_ * 8 * 2) return;
  int bh   = group / 16;
  int rest = group % 16;
  int tm = rest / 2;
  int n0 = (rest % 2) * 64;
  const bf16_t* qb = qh + (size_t)bh * T_ * DKP_;
  const bf16_t* kb = kh + (size_t)bh * T_ * DKP_;
  v8f acc[4];
#pragma unroll
  for (int j = 0; j < 4; ++j) acc[j] = (v8f){0.f,0.f,0.f,0.f,0.f,0.f,0.f,0.f};
  for (int k0 = 0; k0 < DKP_; k0 += 32) {
    v16bf a = load_a16(qb, DKP_, tm*16, k0, lane);
#pragma unroll
    for (int j = 0; j < 4; ++j) {
      v16bf b = load_b_nk(kb, DKP_, k0, n0 + j*16, lane);
      acc[j] = __builtin_amdgcn_wmma_f32_16x16x32_bf16(false, a, false, b,
                                                       (short)0, acc[j], false, false);
    }
  }
  int b   = bh / HEADS_;
  int len = text_len[b];
  int g   = (lane >> 4) & 1;
  float* out = scores + (size_t)bh * T_ * T_;
#pragma unroll
  for (int j = 0; j < 4; ++j) {
    int col = n0 + j*16 + (lane & 15);
#pragma unroll
    for (int r = 0; r < 8; ++r) {
      int row = tm*16 + r + 8*g;
      out[(size_t)row * T_ + col] = (col < len) ? acc[j][r] * 0.1f : NEGV;
    }
  }
}

// ====================== elementwise / staging kernels ======================

__global__ void k_lengths(const int* __restrict__ text, const int* __restrict__ asp,
                          const int* __restrict__ left, int* text_len, int* asp_len,
                          int* left_len) {
  int b = blockIdx.x * blockDim.x + threadIdx.x;
  if (b >= B_) return;
  int c = 0;
  for (int t = 0; t < T_; ++t) c += (text[b*T_+t] != 0);
  text_len[b] = c;
  c = 0;
  for (int l = 0; l < LA_; ++l) c += (asp[b*LA_+l] != 0);
  asp_len[b] = c;
  c = 0;
  for (int l = 0; l < LL_; ++l) c += (left[b*LL_+l] != 0);
  left_len[b] = c;
}

__global__ void convert_pad(const float* __restrict__ src, int R, int C,
                            bf16_t* __restrict__ dst, int Rpad, int Cpad) {
  size_t i = (size_t)blockIdx.x * blockDim.x + threadIdx.x;
  if (i >= (size_t)Rpad * Cpad) return;
  int r = (int)(i / Cpad), c = (int)(i % Cpad);
  float v = (r < R && c < C) ? src[(size_t)r * C + c] : 0.f;
  dst[i] = (bf16_t)v;
}

// dst(CpadRows, RpadCols): dst[c][r] = src[r][c]
__global__ void convert_pad_t(const float* __restrict__ src, int R, int C,
                              bf16_t* __restrict__ dst, int CpadRows, int RpadCols) {
  size_t i = (size_t)blockIdx.x * blockDim.x + threadIdx.x;
  if (i >= (size_t)CpadRows * RpadCols) return;
  int c = (int)(i / RpadCols), r = (int)(i % RpadCols);
  float v = (c < C && r < R) ? src[(size_t)r * C + c] : 0.f;
  dst[i] = (bf16_t)v;
}

__global__ void k_add2(const float* a, const float* b, float* o, int n) {
  int i = blockIdx.x * blockDim.x + threadIdx.x;
  if (i < n) o[i] = a[i] + b[i];
}

// embs (B*T, 352) bf16: [emb | pos_emb | 0pad], with aspect-mean splice
__global__ void k_embs(const int* __restrict__ text, const int* __restrict__ aspect,
                       const int* __restrict__ asp_post, const int* __restrict__ pos_idx,
                       const int* __restrict__ asp_len, const float* __restrict__ emb,
                       const float* __restrict__ pos_emb, bf16_t* __restrict__ out) {
  int bt = blockIdx.x;
  int b = bt / T_, t = bt % T_;
  bool isasp = (t == asp_post[b]);
  float inv = 1.f / (float)asp_len[b];
  int ti = text[bt];
  int pi = pos_idx[bt];
  for (int d = threadIdx.x; d < DINP_; d += blockDim.x) {
    float v = 0.f;
    if (d < DE_) {
      if (isasp) {
        float s = 0.f;
        for (int l = 0; l < LA_; ++l) {
          int ai = aspect[b*LA_ + l];
          if (ai != 0) s += emb[(size_t)ai * DE_ + d];
        }
        v = s * inv;
      } else {
        v = emb[(size_t)ti * DE_ + d];
      }
    } else if (d < DIN_) {
      v = pos_emb[pi * DP_ + (d - DE_)];
    }
    out[(size_t)bt * DINP_ + d] = (bf16_t)v;
  }
}

__global__ void k_lstm_init(bf16_t* h, float* c) {
  int i = blockIdx.x * blockDim.x + threadIdx.x;
  if (i < B_ * HP_) h[i] = (bf16_t)0.f;
  if (i < B_ * H_)  c[i] = 0.f;
}

__global__ void k_lstm_point(const float* __restrict__ gates, bf16_t* __restrict__ h,
                             float* __restrict__ c, float* __restrict__ text_out,
                             const int* __restrict__ text_len, int t, int colOff) {
  int i = blockIdx.x * blockDim.x + threadIdx.x;
  if (i >= B_ * H_) return;
  int b = i / H_, u = i % H_;
  const float* g = gates + (size_t)b * G4_;
  float ig = g[u], fg = g[H_+u], gg = g[2*H_+u], og = g[3*H_+u];
  float si = 1.f / (1.f + __expf(-ig));
  float sf = 1.f / (1.f + __expf(-fg));
  float so = 1.f / (1.f + __expf(-og));
  float cn = sf * c[i] + si * tanhf(gg);
  float hn = so * tanhf(cn);
  bool m = t < text_len[b];
  float outv = 0.f;
  if (m) {
    c[i] = cn;
    h[(size_t)b * HP_ + u] = (bf16_t)hn;
    outv = hn;
  }
  text_out[((size_t)b * T_ + t) * D2_ + colOff + u] = outv;
}

// SE layer: z (B,T,600) f32 -> gated bf16 (B,T,640)
__global__ void k_se(const float* __restrict__ z, const float* __restrict__ w1,
                     const float* __restrict__ b1, const float* __restrict__ w2,
                     const float* __restrict__ b2, const int* __restrict__ text_len,
                     bf16_t* __restrict__ out_bf) {
  __shared__ float m[T_], y[T_], red[T_], y1[16];
  int b = blockIdx.x, t = threadIdx.x;
  const float* zb = z + (size_t)b * T_ * D2_;
  float s = 0.f;
  for (int d = 0; d < D2_; ++d) s += zb[(size_t)t * D2_ + d];
  m[t] = s / (float)D2_;
  __syncthreads();
  if (t < 12) {
    float a = b1[t];
    for (int j = 0; j < T_; ++j) a += m[j] * w1[t*T_ + j];
    y1[t] = fmaxf(a, 0.f);
  }
  __syncthreads();
  float a = b2[t];
  for (int r = 0; r < 12; ++r) a += y1[r] * w2[t*12 + r];
  if (t >= text_len[b]) a += NEGV;
  y[t] = a;
  __syncthreads();
  red[t] = y[t]; __syncthreads();
  for (int off = 64; off > 0; off >>= 1) {
    if (t < off) red[t] = fmaxf(red[t], red[t+off]);
    __syncthreads();
  }
  float mx = red[0]; __syncthreads();
  float e = __expf(y[t] - mx);
  red[t] = e; __syncthreads();
  for (int off = 64; off > 0; off >>= 1) {
    if (t < off) red[t] += red[t+off];
    __syncthreads();
  }
  float p = e / red[0];
  bf16_t* ob = out_bf + (size_t)b * T_ * D2P_;
  for (int d = 0; d < D2P_; ++d) {
    float v = (d < D2_) ? zb[(size_t)t * D2_ + d] * p : 0.f;
    ob[(size_t)t * D2P_ + d] = (bf16_t)v;
  }
}

// pack (B*T,600) f32 -> (B,H,T,128) bf16 per-head, dk padded 100->128
__global__ void k_packqk(const float* __restrict__ qf, bf16_t* __restrict__ qh) {
  size_t i = (size_t)blockIdx.x * blockDim.x + threadIdx.x;
  if (i >= (size_t)B_ * HEADS_ * T_ * DKP_) return;
  int kk = (int)(i % DKP_);
  size_t r = i / DKP_;
  int t = (int)(r % T_);
  size_t r2 = r / T_;
  int h = (int)(r2 % HEADS_);
  int b = (int)(r2 / HEADS_);
  float v = (kk < DK_) ? qf[((size_t)b * T_ + t) * D2_ + h*DK_ + kk] : 0.f;
  qh[i] = (bf16_t)v;
}

// entmax 1.5 per row (128 elems), in place
__global__ void k_entmax(float* __restrict__ scores) {
  __shared__ float xs[T_], s1[T_], s2[T_], tau[T_], red[T_];
  int t = threadIdx.x;
  float* row = scores + (size_t)blockIdx.x * T_;
  float v = row[t] * 0.5f;
  red[t] = v; __syncthreads();
  for (int off = 64; off > 0; off >>= 1) {
    if (t < off) red[t] = fmaxf(red[t], red[t+off]);
    __syncthreads();
  }
  v -= red[0];
  __syncthreads();
  xs[t] = v; __syncthreads();
  for (int k = 2; k <= T_; k <<= 1) {
    for (int j = k >> 1; j > 0; j >>= 1) {
      int ixj = t ^ j;
      if (ixj > t) {
        float a = xs[t], c = xs[ixj];
        bool descBlock = ((t & k) == 0);
        if (descBlock ? (a < c) : (a > c)) { xs[t] = c; xs[ixj] = a; }
      }
      __syncthreads();
    }
  }
  float a1 = xs[t], a2 = xs[t] * xs[t];
  s1[t] = a1; s2[t] = a2; __syncthreads();
  for (int off = 1; off < T_; off <<= 1) {
    float u1 = (t >= off) ? s1[t-off] : 0.f;
    float u2 = (t >= off) ? s2[t-off] : 0.f;
    __syncthreads();
    s1[t] += u1; s2[t] += u2;
    __syncthreads();
  }
  float rho  = (float)(t + 1);
  float mean = s1[t] / rho, msq = s2[t] / rho;
  float ss   = rho * (msq - mean * mean);
  float delta = (1.f - ss) / rho;
  tau[t] = mean - sqrtf(fmaxf(delta, 0.f));
  red[t] = (tau[t] <= xs[t]) ? 1.f : 0.f;
  __syncthreads();
  for (int off = 64; off > 0; off >>= 1) {
    if (t < off) red[t] += red[t+off];
    __syncthreads();
  }
  int support = (int)(red[0] + 0.5f);
  float taus = tau[support - 1];
  float d = fmaxf(v - taus, 0.f);
  row[t] = d * d;
}

__global__ void k_headavg(const float* __restrict__ probs,
                          const int* __restrict__ text_len,
                          float* __restrict__ adj_sem) {
  size_t i = (size_t)blockIdx.x * blockDim.x + threadIdx.x;
  if (i >= (size_t)B_ * T_ * T_) return;
  int j = (int)(i % T_);
  size_t r = i / T_;
  int ii = (int)(r % T_);
  int b  = (int)(r / T_);
  float s = 0.f;
  for (int h = 0; h < HEADS_; ++h)
    s += probs[(((size_t)b * HEADS_ + h) * T_ + ii) * T_ + j];
  float v = s / (float)HEADS_;
  if (ii == j) v = 1.f;
  if (ii >= text_len[b]) v = 0.f;
  adj_sem[i] = v;
}

__global__ void k_deg(const float* __restrict__ A, float* __restrict__ dn) {
  int i = blockIdx.x * blockDim.x + threadIdx.x;
  if (i >= B_ * T_) return;
  const float* row = A + (size_t)i * T_;
  float s = 0.f;
  for (int j = 0; j < T_; ++j) s += row[j];
  dn[i] = rsqrtf(s + 1.f);
}

__global__ void k_norm(const float* __restrict__ A, const float* __restrict__ dn,
                       bf16_t* __restrict__ An) {
  size_t i = (size_t)blockIdx.x * blockDim.x + threadIdx.x;
  if (i >= (size_t)B_ * T_ * T_) return;
  int j = (int)(i % T_);
  size_t r = i / T_;
  int ii = (int)(r % T_);
  int b  = (int)(r / T_);
  An[i] = (bf16_t)(dn[b*T_ + ii] * A[i] * dn[b*T_ + j]);
}

// biaffine attention for the single pooled row + final FC
__global__ void k_pool_fc(const float* __restrict__ x, const float* __restrict__ x_sem,
                          const float* __restrict__ aff1, const float* __restrict__ aff2,
                          const float* __restrict__ fcw, const float* __restrict__ fcb,
                          const int* __restrict__ left_len, float* __restrict__ out) {
  __shared__ float xr[D2_], xsr[D2_], v[D2_], lg[T_], red[T_], pooled[2*D2_];
  int b = blockIdx.x, t = threadIdx.x;
  int istar = left_len[b];
  const float* xrow  = x     + ((size_t)b * T_ + istar) * D2_;
  const float* xsrow = x_sem + ((size_t)b * T_ + istar) * D2_;
  for (int d = t; d < D2_; d += T_) { xr[d] = xrow[d]; xsr[d] = xsrow[d]; }
  __syncthreads();
  // ---- branch 1: A1 = softmax((x@aff1) . x_sem^T)[istar] ; pooled1 = A1 @ x_sem
  for (int d = t; d < D2_; d += T_) {
    float s = 0.f;
    for (int k = 0; k < D2_; ++k) s += xr[k] * aff1[(size_t)k * D2_ + d];
    v[d] = s;
  }
  __syncthreads();
  {
    const float* rowj = x_sem + ((size_t)b * T_ + t) * D2_;
    float s = 0.f;
    for (int d = 0; d < D2_; ++d) s += v[d] * rowj[d];
    lg[t] = s;
  }
  __syncthreads();
  red[t] = lg[t]; __syncthreads();
  for (int off = 64; off > 0; off >>= 1) { if (t < off) red[t] = fmaxf(red[t], red[t+off]); __syncthreads(); }
  float mx = red[0]; __syncthreads();
  float e = __expf(lg[t] - mx);
  red[t] = e; __syncthreads();
  for (int off = 64; off > 0; off >>= 1) { if (t < off) red[t] += red[t+off]; __syncthreads(); }
  float p = e / red[0];
  __syncthreads();
  lg[t] = p;
  __syncthreads();
  for (int d = t; d < D2_; d += T_) {
    float s = 0.f;
    for (int j = 0; j < T_; ++j) s += lg[j] * x_sem[((size_t)b * T_ + j) * D2_ + d];
    pooled[d] = s;
  }
  __syncthreads();
  // ---- branch 2: A2 = softmax((x_sem@aff2) . x^T)[istar] ; pooled2 = A2 @ x
  for (int d = t; d < D2_; d += T_) {
    float s = 0.f;
    for (int k = 0; k < D2_; ++k) s += xsr[k] * aff2[(size_t)k * D2_ + d];
    v[d] = s;
  }
  __syncthreads();
  {
    const float* rowj = x + ((size_t)b * T_ + t) * D2_;
    float s = 0.f;
    for (int d = 0; d < D2_; ++d) s += v[d] * rowj[d];
    lg[t] = s;
  }
  __syncthreads();
  red[t] = lg[t]; __syncthreads();
  for (int off = 64; off > 0; off >>= 1) { if (t < off) red[t] = fmaxf(red[t], red[t+off]); __syncthreads(); }
  float mx2 = red[0]; __syncthreads();
  float e2 = __expf(lg[t] - mx2);
  red[t] = e2; __syncthreads();
  for (int off = 64; off > 0; off >>= 1) { if (t < off) red[t] += red[t+off]; __syncthreads(); }
  float p2 = e2 / red[0];
  __syncthreads();
  lg[t] = p2;
  __syncthreads();
  for (int d = t; d < D2_; d += T_) {
    float s = 0.f;
    for (int j = 0; j < T_; ++j) s += lg[j] * x[((size_t)b * T_ + j) * D2_ + d];
    pooled[D2_ + d] = s;
  }
  __syncthreads();
  if (t < POL_) {
    float s = fcb[t];
    for (int d = 0; d < 2*D2_; ++d) s += pooled[d] * fcw[d * POL_ + t];
    out[b * POL_ + t] = s;
  }
}

// ====================== host orchestration ======================

static inline size_t alignUp256(size_t x) { return (x + 255) & ~(size_t)255; }
static inline int cdiv(long long a, long long b) { return (int)((a + b - 1) / b); }

extern "C" void kernel_launch(void* const* d_in, const int* in_sizes, int n_in,
                              void* d_out, int out_size, void* d_ws, size_t ws_size,
                              hipStream_t stream) {
  (void)in_sizes; (void)n_in; (void)out_size; (void)ws_size;

  const int*   text_idx = (const int*)d_in[0];
  const int*   asp_idx  = (const int*)d_in[1];
  const int*   left_idx = (const int*)d_in[2];
  const int*   pos_idx  = (const int*)d_in[3];
  const int*   asp_post = (const int*)d_in[4];
  const float* adj      = (const float*)d_in[6];
  const float* emb      = (const float*)d_in[7];
  const float* pos_emb  = (const float*)d_in[8];
  const float* WihF = (const float*)d_in[9],  *WhhF = (const float*)d_in[10];
  const float* bihF = (const float*)d_in[11], *bhhF = (const float*)d_in[12];
  const float* WihB = (const float*)d_in[13], *WhhB = (const float*)d_in[14];
  const float* bihB = (const float*)d_in[15], *bhhB = (const float*)d_in[16];
  const float* wq = (const float*)d_in[17], *bq = (const float*)d_in[18];
  const float* wk = (const float*)d_in[19], *bk = (const float*)d_in[20];
  const float* se_w1 = (const float*)d_in[21], *se_b1 = (const float*)d_in[22];
  const float* se_w2 = (const float*)d_in[23], *se_b2 = (const float*)d_in[24];
  const float* gc1_w = (const float*)d_in[25], *gc1_b = (const float*)d_in[26];
  const float* gc2_w = (const float*)d_in[27], *gc2_b = (const float*)d_in[28];
  const float* aff1 = (const float*)d_in[29], *aff2 = (const float*)d_in[30];
  const float* fcw = (const float*)d_in[31], *fcb = (const float*)d_in[32];
  float* outp = (float*)d_out;

  // ---- workspace bump allocator
  char* base = (char*)d_ws;
  size_t off = 0;
  auto alloc = [&](size_t bytes) -> void* {
    void* r = base + off;
    off = alignUp256(off + bytes);
    return r;
  };

  int* text_len = (int*)alloc(B_ * sizeof(int));
  int* asp_len  = (int*)alloc(B_ * sizeof(int));
  int* left_len = (int*)alloc(B_ * sizeof(int));

  bf16_t* embs_bf = (bf16_t*)alloc((size_t)B_*T_*DINP_ * 2);
  bf16_t* WihF_bf = (bf16_t*)alloc((size_t)G4P_*DINP_ * 2);
  bf16_t* WhhF_bf = (bf16_t*)alloc((size_t)G4P_*HP_ * 2);
  bf16_t* WihB_bf = (bf16_t*)alloc((size_t)G4P_*DINP_ * 2);
  bf16_t* WhhB_bf = (bf16_t*)alloc((size_t)G4P_*HP_ * 2);
  bf16_t* wq_bf   = (bf16_t*)alloc((size_t)D2P_*D2P_ * 2);
  bf16_t* wk_bf   = (bf16_t*)alloc((size_t)D2P_*D2P_ * 2);
  bf16_t* gc1T_bf = (bf16_t*)alloc((size_t)D2P_*D2P_ * 2);
  bf16_t* gc2T_bf = (bf16_t*)alloc((size_t)D2P_*D2P_ * 2);
  float*  biasF   = (float*)alloc(G4_ * 4);
  float*  biasB   = (float*)alloc(G4_ * 4);

  float* gatesX = (float*)alloc((size_t)B_*T_*G4_ * 4);   // reused as qf/kf later
  float* qf = gatesX;
  float* kf = gatesX + (size_t)B_*T_*D2_;

  bf16_t* h_bf  = (bf16_t*)alloc((size_t)B_*HP_ * 2);
  float*  c_f   = (float*)alloc((size_t)B_*H_ * 4);
  float*  gstep = (float*)alloc((size_t)B_*G4_ * 4);

  float*  text_out    = (float*)alloc((size_t)B_*T_*D2_ * 4);
  bf16_t* text_out_bf = (bf16_t*)alloc((size_t)B_*T_*D2P_ * 2);

  bf16_t* qh = (bf16_t*)alloc((size_t)B_*HEADS_*T_*DKP_ * 2);
  bf16_t* kh = (bf16_t*)alloc((size_t)B_*HEADS_*T_*DKP_ * 2);

  float* scores = (float*)alloc((size_t)B_*HEADS_*T_*T_ * 4);  // reused as x later
  float* x_f = scores;

  float*  adj_sem = (float*)alloc((size_t)B_*T_*T_ * 4);
  float*  dn_sem  = (float*)alloc((size_t)B_*T_ * 4);
  float*  dn_adj  = (float*)alloc((size_t)B_*T_ * 4);
  bf16_t* AnSem   = (bf16_t*)alloc((size_t)B_*T_*T_ * 2);
  bf16_t* AnAdj   = (bf16_t*)alloc((size_t)B_*T_*T_ * 2);

  bf16_t* se_bf  = (bf16_t*)alloc((size_t)B_*T_*D2P_ * 2);   // se1, then se2
  float*  tmp    = (float*)alloc((size_t)B_*T_*D2_ * 4);     // gcn stage-A out (x2)
  bf16_t* tmp_bf = (bf16_t*)alloc((size_t)B_*T_*D2P_ * 2);
  float*  x_sem  = (float*)alloc((size_t)B_*T_*D2_ * 4);

  // ---- 0) lengths
  k_lengths<<<1, 128, 0, stream>>>(text_idx, asp_idx, left_idx, text_len, asp_len, left_len);

  // ---- 1) weight staging (bf16, N padded to 64-multiples, K to 32-multiples)
  convert_pad<<<cdiv((long long)G4P_*DINP_,256),256,0,stream>>>(WihF, G4_, DIN_, WihF_bf, G4P_, DINP_);
  convert_pad<<<cdiv((long long)G4P_*HP_  ,256),256,0,stream>>>(WhhF, G4_, H_,   WhhF_bf, G4P_, HP_);
  convert_pad<<<cdiv((long long)G4P_*DINP_,256),256,0,stream>>>(WihB, G4_, DIN_, WihB_bf, G4P_, DINP_);
  convert_pad<<<cdiv((long long)G4P_*HP_  ,256),256,0,stream>>>(WhhB, G4_, H_,   WhhB_bf, G4P_, HP_);
  convert_pad<<<cdiv((long long)D2P_*D2P_,256),256,0,stream>>>(wq, D2_, D2_, wq_bf, D2P_, D2P_);
  convert_pad<<<cdiv((long long)D2P_*D2P_,256),256,0,stream>>>(wk, D2_, D2_, wk_bf, D2P_, D2P_);
  convert_pad_t<<<cdiv((long long)D2P_*D2P_,256),256,0,stream>>>(gc1_w, D2_, D2_, gc1T_bf, D2P_, D2P_);
  convert_pad_t<<<cdiv((long long)D2P_*D2P_,256),256,0,stream>>>(gc2_w, D2_, D2_, gc2T_bf, D2P_, D2P_);
  k_add2<<<cdiv(G4_,256),256,0,stream>>>(bihF, bhhF, biasF, G4_);
  k_add2<<<cdiv(G4_,256),256,0,stream>>>(bihB, bhhB, biasB, G4_);

  // ---- 2) embeddings + aspect splice -> bf16 (B*T, 352)
  k_embs<<<B_*T_, 128, 0, stream>>>(text_idx, asp_idx, asp_post, pos_idx, asp_len,
                                    emb, pos_emb, embs_bf);

  const int BT = B_ * T_;
  // ---- 3) BiLSTM: big input-projection WMMA GEMM, then sequential recurrence
  {
    int groups = (BT/16) * (G4P_/64);
    gemm_nk<<<cdiv(groups,4),128,0,stream>>>(embs_bf, DINP_, WihF_bf, DINP_,
                                             gatesX, G4_, biasF, nullptr, 0,
                                             BT, G4P_, G4_, DINP_, 0);
    k_lstm_init<<<cdiv(B_*HP_,256),256,0,stream>>>(h_bf, c_f);
    int sgroups = (B_/16) * (G4P_/64);
    for (int t = 0; t < T_; ++t) {
      gemm_nk<<<cdiv(sgroups,4),128,0,stream>>>(h_bf, HP_, WhhF_bf, HP_,
                                                gstep, G4_, nullptr,
                                                gatesX + (size_t)t*G4_, T_*G4_,
                                                B_, G4P_, G4_, HP_, 0);
      k_lstm_point<<<cdiv(B_*H_,256),256,0,stream>>>(gstep, h_bf, c_f, text_out,
                                                     text_len, t, 0);
    }
    gemm_nk<<<cdiv(groups,4),128,0,stream>>>(embs_bf, DINP_, WihB_bf, DINP_,
                                             gatesX, G4_, biasB, nullptr, 0,
                                             BT, G4P_, G4_, DINP_, 0);
    k_lstm_init<<<cdiv(B_*HP_,256),256,0,stream>>>(h_bf, c_f);
    for (int t = T_ - 1; t >= 0; --t) {
      gemm_nk<<<cdiv(sgroups,4),128,0,stream>>>(h_bf, HP_, WhhB_bf, HP_,
                                                gstep, G4_, nullptr,
                                                gatesX + (size_t)t*G4_, T_*G4_,
                                                B_, G4P_, G4_, HP_, 0);
      k_lstm_point<<<cdiv(B_*H_,256),256,0,stream>>>(gstep, h_bf, c_f, text_out,
                                                     text_len, t, H_);
    }
  }

  // ---- 4) Q/K projections (gatesX is free now -> reused as qf/kf)
  convert_pad<<<cdiv((long long)BT*D2P_,256),256,0,stream>>>(text_out, BT, D2_,
                                                             text_out_bf, BT, D2P_);
  {
    int groups = (BT/16) * (D2P_/64);
    gemm_nk<<<cdiv(groups,4),128,0,stream>>>(text_out_bf, D2P_, wq_bf, D2P_,
                                             qf, D2_, bq, nullptr, 0,
                                             BT, D2P_, D2_, D2P_, 0);
    gemm_nk<<<cdiv(groups,4),128,0,stream>>>(text_out_bf, D2P_, wk_bf, D2P_,
                                             kf, D2_, bk, nullptr, 0,
                                             BT, D2P_, D2_, D2P_, 0);
  }
  {
    long long tot = (long long)B_*HEADS_*T_*DKP_;
    k_packqk<<<cdiv(tot,256),256,0,stream>>>(qf, qh);
    k_packqk<<<cdiv(tot,256),256,0,stream>>>(kf, kh);
  }

  // ---- 5) attention scores + entmax1.5 + head average
  scores_gemm<<<cdiv((long long)B_*HEADS_*16,4),128,0,stream>>>(qh, kh, scores, text_len);
  k_entmax<<<B_*HEADS_*T_, 128, 0, stream>>>(scores);
  k_headavg<<<cdiv((long long)B_*T_*T_,256),256,0,stream>>>(scores, text_len, adj_sem);

  // ---- 6) adjacency normalization (semantic + syntactic)
  k_deg<<<cdiv(B_*T_,256),256,0,stream>>>(adj_sem, dn_sem);
  k_norm<<<cdiv((long long)B_*T_*T_,256),256,0,stream>>>(adj_sem, dn_sem, AnSem);
  k_deg<<<cdiv(B_*T_,256),256,0,stream>>>(adj, dn_adj);
  k_norm<<<cdiv((long long)B_*T_*T_,256),256,0,stream>>>(adj, dn_adj, AnAdj);

  // ---- 7) GCN layer 1: x_sem = relu((AnSem @ se(text_out)) @ gc1_w + b)
  k_se<<<B_, 128, 0, stream>>>(text_out, se_w1, se_b1, se_w2, se_b2, text_len, se_bf);
  {
    long long groups = (long long)B_ * (T_/16) * (D2P_/64);
    gemm_kn_batched<<<cdiv(groups,4),128,0,stream>>>(AnSem, T_, (long long)T_*T_,
                                                     se_bf, D2P_, (long long)T_*D2P_,
                                                     tmp, D2_, (long long)T_*D2_,
                                                     D2P_, D2_, T_);
  }
  convert_pad<<<cdiv((long long)BT*D2P_,256),256,0,stream>>>(tmp, BT, D2_, tmp_bf, BT, D2P_);
  {
    int groups = (BT/16) * (D2P_/64);
    gemm_nk<<<cdiv(groups,4),128,0,stream>>>(tmp_bf, D2P_, gc1T_bf, D2P_,
                                             x_sem, D2_, gc1_b, nullptr, 0,
                                             BT, D2P_, D2_, D2P_, 1);
  }

  // ---- 8) GCN layer 2: x = relu((AnAdj @ se(x_sem)) @ gc2_w + b)  (x aliases scores)
  k_se<<<B_, 128, 0, stream>>>(x_sem, se_w1, se_b1, se_w2, se_b2, text_len, se_bf);
  {
    long long groups = (long long)B_ * (T_/16) * (D2P_/64);
    gemm_kn_batched<<<cdiv(groups,4),128,0,stream>>>(AnAdj, T_, (long long)T_*T_,
                                                     se_bf, D2P_, (long long)T_*D2P_,
                                                     tmp, D2_, (long long)T_*D2_,
                                                     D2P_, D2_, T_);
  }
  convert_pad<<<cdiv((long long)BT*D2P_,256),256,0,stream>>>(tmp, BT, D2_, tmp_bf, BT, D2P_);
  {
    int groups = (BT/16) * (D2P_/64);
    gemm_nk<<<cdiv(groups,4),128,0,stream>>>(tmp_bf, D2P_, gc2T_bf, D2P_,
                                             x_f, D2_, gc2_b, nullptr, 0,
                                             BT, D2P_, D2_, D2P_, 1);
  }

  // ---- 9) biaffine attention (only the pooled row is needed) + final FC
  k_pool_fc<<<B_, 128, 0, stream>>>(x_f, x_sem, aff1, aff2, fcw, fcb, left_len, outp);
}
